// NeuralHawkes_46832323396297
// MI455X (gfx1250) — compile-verified, matrix operand order checked by reading, not run on hardware
//
#include <hip/hip_runtime.h>
#include <hip/hip_bf16.h>

typedef __attribute__((ext_vector_type(16))) __bf16 v16bf;
typedef __attribute__((ext_vector_type(8)))  float  v8f;

union FragA { v16bf v; unsigned int u[8]; };
union FragB { v16bf v; uint4 q[2]; };

#define BN    64
#define TP1   1023
#define TP2   1024
#define HD    256
#define TWOH  512
#define SEVENH 1792
#define KCLS  128
#define SN    1024
#define EPSF  2.220446049250313e-16f

__device__ __forceinline__ float sigf(float x)  { return 1.f / (1.f + __expf(-x)); }
__device__ __forceinline__ float splusf(float x){ return (x > 20.f) ? x : __logf(1.f + __expf(x)); }

// ---------------------------------------------------------------------------
// Kernel Z: zero the 3 per-batch accumulators (192 floats)
// ---------------------------------------------------------------------------
__global__ void zero_kernel(float* p, int n) {
    int i = blockIdx.x * blockDim.x + threadIdx.x;
    if (i < n) p[i] = 0.f;
}

// ---------------------------------------------------------------------------
// Kernel A: pre-swizzle W (1792x512) and Wl (128x256) into WMMA B-fragment
// layout (per-lane-contiguous 16 bf16 = 32B), convert Emb to bf16.
// B-fragment element map (wave32, 16-bit B 32x16): lane l -> column n=l&15,
// khalf = (l<16)?0:8 ; element e in [0,16): K = e + (e>=8?8:0) + khalf.
// ---------------------------------------------------------------------------
__global__ __launch_bounds__(256) void pack_kernel(
    const float* __restrict__ W, const float* __restrict__ Wl,
    const float* __restrict__ Emb,
    __bf16* __restrict__ Wpack, __bf16* __restrict__ Wlpack,
    __bf16* __restrict__ EmbBf)
{
    const int NW  = SEVENH * TWOH;   // 917504 : 16 ktile x 112 jtile x 512
    const int NWL = HD * KCLS;       // 32768  : 8 ktile x 8 jtile x 512
    const int NE  = 131 * HD;        // 33536
    const int total = NW + NWL + NE;
    for (int q = blockIdx.x * blockDim.x + threadIdx.x; q < total;
         q += gridDim.x * blockDim.x) {
        if (q < NW) {
            int tile = q >> 9, r = q & 511;
            int lane = r >> 4, e = r & 15;
            int kt = tile / 112, jt = tile % 112;
            int n = lane & 15;
            int khalf = (lane < 16) ? 0 : 8;
            int K = e + ((e >= 8) ? 8 : 0) + khalf;
            Wpack[q] = (__bf16)W[(jt * 16 + n) * TWOH + (kt * 32 + K)];
        } else if (q < NW + NWL) {
            int q2 = q - NW;
            int tile = q2 >> 9, r = q2 & 511;
            int lane = r >> 4, e = r & 15;
            int kt = tile >> 3, jt = tile & 7;
            int n = lane & 15;
            int khalf = (lane < 16) ? 0 : 8;
            int K = e + ((e >= 8) ? 8 : 0) + khalf;
            Wlpack[q2] = (__bf16)Wl[(jt * 16 + n) * HD + (kt * 32 + K)];
        } else {
            int q3 = q - NW - NWL;
            EmbBf[q3] = (__bf16)Emb[q3];
        }
    }
}

// ---------------------------------------------------------------------------
// Kernel B: CT-LSTM scan. 4 workgroups x 16 sequences, 16 waves (512 thr).
// Per step: z(16x1792) = x(16x512) @ Wpacked via 112 bf16 WMMAs per wave;
// wave w owns hidden units [16w,16w+16) across all 7 gate chunks, so the
// c/cbar state for its (row, hidden) cells lives in VGPRs for all 1023 steps.
// ---------------------------------------------------------------------------
__global__ __launch_bounds__(512) void scan_kernel(
    const int*   __restrict__ event,
    const float* __restrict__ dtime,
    const __bf16* __restrict__ EmbBf,
    const __bf16* __restrict__ Wpack,
    const float* __restrict__ bias,
    float* __restrict__ all_c, float* __restrict__ all_cb,
    float* __restrict__ all_gd, float* __restrict__ all_go,
    float* __restrict__ all_h)
{
    __shared__ __align__(16) __bf16 x_lds[16][TWOH];
    __shared__ float dts_lds[16];

    const int wg   = blockIdx.x;          // 0..3 -> sequences wg*16..wg*16+15
    const int tid  = threadIdx.x;
    const int wave = tid >> 5;            // 0..15 -> hidden slice
    const int lane = tid & 31;
    const int nloc = lane & 15;
    const int khalf = (lane < 16) ? 0 : 8;
    const int rbase = (lane < 16) ? 0 : 8;
    const int hid  = wave * 16 + nloc;    // hidden unit owned by this lane

    // zero h part of x and states
    for (int i = tid; i < 16 * HD; i += 512)
        x_lds[i / HD][HD + (i % HD)] = (__bf16)0.f;

    float cst[8], cbst[8];
#pragma unroll
    for (int r = 0; r < 8; ++r) { cst[r] = 0.f; cbst[r] = 0.f; }

    float bvals[7];
#pragma unroll
    for (int c = 0; c < 7; ++c) bvals[c] = bias[c * HD + hid];

    for (int t = 0; t < TP1; ++t) {
        // ---- stage embeddings for this step: x_lds[:, 0:256] ----
        {
            int row = tid >> 5;                 // 16 rows, 32 threads each
            int colbase = (tid & 31) * 8;
            int b = wg * 16 + row;
            int ev = event[b * TP2 + t];
            const __bf16* src = EmbBf + (size_t)ev * HD + colbase;
#pragma unroll
            for (int k = 0; k < 8; ++k) x_lds[row][colbase + k] = src[k];
            if (tid < 16) dts_lds[tid] = dtime[(wg * 16 + tid) * TP2 + t + 1];
        }
        __syncthreads();

        float dtv[8];
#pragma unroll
        for (int r = 0; r < 8; ++r) dtv[r] = dts_lds[rbase + r];

        // ---- GEMM: 7 N-tiles per wave, K=512 in 16 slices of 32 ----
        v8f acc[7];
#pragma unroll
        for (int c = 0; c < 7; ++c) {
            float bv = bvals[c];
            acc[c] = (v8f){bv, bv, bv, bv, bv, bv, bv, bv};
        }
        for (int kt = 0; kt < 16; ++kt) {
            FragA a;
            const int m = lane & 15;
#pragma unroll
            for (int p = 0; p < 8; ++p) {
                int K0 = 2 * p + ((p >= 4) ? 8 : 0) + khalf;
                a.u[p] = *(const unsigned int*)&x_lds[m][kt * 32 + K0];
            }
#pragma unroll
            for (int c = 0; c < 7; ++c) {
                int jt = c * 16 + wave;
                FragB bf;
                const uint4* bp = (const uint4*)(Wpack +
                    ((size_t)(kt * 112 + jt) * 32 + lane) * 16);
                bf.q[0] = bp[0]; bf.q[1] = bp[1];
                acc[c] = __builtin_amdgcn_wmma_f32_16x16x32_bf16(
                    false, a.v, false, bf.v, (short)0, acc[c], false, false);
            }
        }
        __syncthreads();   // all A-reads of x_lds done

        // ---- gates + CT-LSTM update, state in registers ----
        float hnew[8];
#pragma unroll
        for (int r = 0; r < 8; ++r) {
            int i = rbase + r;
            int b = wg * 16 + i;
            float gi  = sigf(acc[0][r]);
            float gf  = sigf(acc[1][r]);
            float go  = sigf(acc[2][r]);
            float zc  = tanhf(acc[3][r]);
            float gib = sigf(acc[4][r]);
            float gfb = sigf(acc[5][r]);
            float gd  = splusf(acc[6][r]);
            float ci  = gf * cst[r] + gi * zc;
            float cbi = gfb * cbst[r] + gib * zc;
            float cn  = cbi + (ci - cbi) * __expf(-gd * dtv[r]);
            float hn  = go * tanhf(cn);
            cst[r] = cn; cbst[r] = cbi;
            size_t ob = ((size_t)b * TP1 + t) * HD + hid;
            all_c[ob]  = ci;
            all_cb[ob] = cbi;
            all_gd[ob] = gd;
            all_go[ob] = go;
            all_h[ob]  = hn;
            hnew[r] = hn;
        }
        // feed h back for next step (cols 256..511, disjoint from e writes)
#pragma unroll
        for (int r = 0; r < 8; ++r)
            x_lds[rbase + r][HD + hid] = (__bf16)hnew[r];
    }
}

// ---------------------------------------------------------------------------
// Kernel C: sum_t log(softplus(h_t . Wl[tgt]) + EPS) * mask  per batch.
// Tiny FLOP count -> plain VALU dot + block reduction.
// ---------------------------------------------------------------------------
__global__ __launch_bounds__(256) void loglam_kernel(
    const int* __restrict__ event, const float* __restrict__ all_h,
    const float* __restrict__ Wl, float* __restrict__ loglam_sum)
{
    __shared__ float red[256];
    const int b = blockIdx.x, tid = threadIdx.x;
    float local = 0.f;
    for (int t = tid; t < TP1; t += 256) {
        int traw = event[b * TP2 + t + 1];
        float mask = (traw < KCLS) ? 1.f : 0.f;
        int tgt = (traw < KCLS) ? traw : 0;
        const float4* h4 = (const float4*)(all_h + ((size_t)b * TP1 + t) * HD);
        const float4* w4 = (const float4*)(Wl + (size_t)tgt * HD);
        float dot = 0.f;
#pragma unroll 4
        for (int k = 0; k < HD / 4; ++k) {
            float4 a = h4[k], w = w4[k];
            dot += a.x * w.x + a.y * w.y + a.z * w.z + a.w * w.w;
        }
        local += __logf(splusf(dot) + EPSF) * mask;
    }
    red[tid] = local;
    __syncthreads();
    for (int s = 128; s > 0; s >>= 1) {
        if (tid < s) red[tid] += red[tid + s];
        __syncthreads();
    }
    if (tid == 0) loglam_sum[b] = red[0];
}

// ---------------------------------------------------------------------------
// Kernel D: MC-integral sampling. Each wave owns a 16-sample tile:
// gather states, hy = go*tanh(cb+(c-cb)exp(-gd*dt)) -> bf16 in LDS,
// (16x256)@(256x128) via 64 WMMAs, softplus + cross-lane reduce, atomics.
// ---------------------------------------------------------------------------
__global__ __launch_bounds__(128) void sample_kernel(
    const int*   __restrict__ idxsamp,
    const float* __restrict__ dts_samp,
    const float* __restrict__ mask_samp,
    const float* __restrict__ all_c, const float* __restrict__ all_cb,
    const float* __restrict__ all_gd, const float* __restrict__ all_go,
    const __bf16* __restrict__ Wlpack,
    float* __restrict__ lamsum, float* __restrict__ masksum)
{
    __shared__ __align__(16) __bf16 hy[4][16][HD];
    const int wave = threadIdx.x >> 5;
    const int lane = threadIdx.x & 31;
    const int khalf = (lane < 16) ? 0 : 8;
    const int rbase = (lane < 16) ? 0 : 8;

    for (int iter = 0; iter < 4; ++iter) {
        int T  = iter * 1024 + blockIdx.x * 4 + wave;   // 0..4095 sample tiles
        int b  = T >> 6;                                // 64 tiles per batch
        int s0 = (T & 63) * 16;

        // build 16 sample rows of hy (this wave's private LDS slab)
        for (int sl = 0; sl < 16; ++sl) {
            int sidx = b * SN + s0 + sl;
            size_t row = (size_t)idxsamp[sidx] * HD;
            float dts = dts_samp[sidx];
            int h0 = lane * 8;
#pragma unroll
            for (int k = 0; k < 8; ++k) {
                int h = h0 + k;
                float cv  = all_c[row + h];
                float cbv = all_cb[row + h];
                float gdv = all_gd[row + h];
                float gov = all_go[row + h];
                float cs  = cbv + (cv - cbv) * __expf(-gdv * dts);
                hy[wave][sl][h] = (__bf16)(gov * tanhf(cs));
            }
        }
        // same-wave LDS ops are in-order: WMMA A-reads see the stores.

        v8f acc[8];
#pragma unroll
        for (int jt = 0; jt < 8; ++jt) acc[jt] = (v8f){0,0,0,0,0,0,0,0};
        for (int kt = 0; kt < 8; ++kt) {
            FragA a;
            const int m = lane & 15;
#pragma unroll
            for (int p = 0; p < 8; ++p) {
                int K0 = 2 * p + ((p >= 4) ? 8 : 0) + khalf;
                a.u[p] = *(const unsigned int*)&hy[wave][m][kt * 32 + K0];
            }
#pragma unroll
            for (int jt = 0; jt < 8; ++jt) {
                FragB bf;
                const uint4* bp = (const uint4*)(Wlpack +
                    ((size_t)(kt * 8 + jt) * 32 + lane) * 16);
                bf.q[0] = bp[0]; bf.q[1] = bp[1];
                acc[jt] = __builtin_amdgcn_wmma_f32_16x16x32_bf16(
                    false, a.v, false, bf.v, (short)0, acc[jt], false, false);
            }
        }

        // softplus over all 128 classes, reduce over N per sample row
        float rowsum[8];
#pragma unroll
        for (int r = 0; r < 8; ++r) rowsum[r] = 0.f;
#pragma unroll
        for (int jt = 0; jt < 8; ++jt)
#pragma unroll
            for (int r = 0; r < 8; ++r) rowsum[r] += splusf(acc[jt][r]);
#pragma unroll
        for (int off = 8; off > 0; off >>= 1)
#pragma unroll
            for (int r = 0; r < 8; ++r)
                rowsum[r] += __shfl_xor(rowsum[r], off, 32);

        if ((lane & 15) == 0) {   // lanes 0 (rows 0-7) and 16 (rows 8-15)
            float lsum = 0.f, msum = 0.f;
#pragma unroll
            for (int r = 0; r < 8; ++r) {
                int s = s0 + rbase + r;
                float mk = mask_samp[b * SN + s];
                lsum += rowsum[r] * mk;
                msum += mk;
            }
            atomicAdd(&lamsum[b], lsum);
            atomicAdd(&masksum[b], msum);
        }
    }
}

// ---------------------------------------------------------------------------
// Kernel E: loglik[b] = loglam_sum[b] - (lamsum[b]/masksum[b]) * duration[b]
// ---------------------------------------------------------------------------
__global__ void finalize_kernel(const float* loglam_sum, const float* lamsum,
                                const float* masksum, const float* duration,
                                float* out)
{
    int b = threadIdx.x;
    if (b < BN)
        out[b] = loglam_sum[b] - (lamsum[b] / masksum[b]) * duration[b];
}

// ---------------------------------------------------------------------------
extern "C" void kernel_launch(void* const* d_in, const int* in_sizes, int n_in,
                              void* d_out, int out_size, void* d_ws, size_t ws_size,
                              hipStream_t stream) {
    const int*   event    = (const int*)d_in[0];
    const float* dtime    = (const float*)d_in[1];
    /* d_in[2] = post (unused by reference) */
    const float* duration = (const float*)d_in[3];
    const float* dts_samp = (const float*)d_in[4];
    const int*   idxsamp  = (const int*)d_in[5];
    const float* msk_samp = (const float*)d_in[6];
    const float* Emb      = (const float*)d_in[7];
    const float* W        = (const float*)d_in[8];
    const float* bias     = (const float*)d_in[9];
    const float* Wl       = (const float*)d_in[10];
    float* out = (float*)d_out;

    // workspace carve-up (256B-aligned slabs)
    char* ws = (char*)d_ws;
    size_t off = 0;
    auto take = [&](size_t bytes) {
        char* p = ws + off;
        off += (bytes + 255) & ~(size_t)255;
        return p;
    };
    __bf16* Wpack  = (__bf16*)take((size_t)SEVENH * TWOH * 2);   // 1.84 MB
    __bf16* Wlpack = (__bf16*)take((size_t)HD * KCLS * 2);       // 64 KB
    __bf16* EmbBf  = (__bf16*)take((size_t)131 * HD * 2);        // 67 KB
    const size_t state_elems = (size_t)BN * TP1 * HD;            // 16.76 M
    float* all_c  = (float*)take(state_elems * 4);
    float* all_cb = (float*)take(state_elems * 4);
    float* all_gd = (float*)take(state_elems * 4);
    float* all_go = (float*)take(state_elems * 4);
    float* all_h  = (float*)take(state_elems * 4);
    float* accums = (float*)take(192 * 4);
    float* loglam_sum = accums;
    float* lamsum     = accums + 64;
    float* masksum    = accums + 128;

    zero_kernel<<<1, 192, 0, stream>>>(accums, 192);
    pack_kernel<<<1024, 256, 0, stream>>>(W, Wl, Emb, Wpack, Wlpack, EmbBf);
    scan_kernel<<<4, 512, 0, stream>>>(event, dtime, EmbBf, Wpack, bias,
                                       all_c, all_cb, all_gd, all_go, all_h);
    loglam_kernel<<<BN, 256, 0, stream>>>(event, all_h, Wl, loglam_sum);
    sample_kernel<<<256, 128, 0, stream>>>(idxsamp, dts_samp, msk_samp,
                                           all_c, all_cb, all_gd, all_go,
                                           Wlpack, lamsum, masksum);
    finalize_kernel<<<1, 64, 0, stream>>>(loglam_sum, lamsum, masksum,
                                          duration, out);
}
